// RCCA_90615220011619
// MI455X (gfx1250) — matrix operand order repaired and checked by reading, not compiled, not run on hardware
//
#include <hip/hip_runtime.h>
#include <cmath>

// ---------------------------------------------------------------------------
// RCCA (recurrent criss-cross attention) for MI455X / gfx1250.
// All contractions run through V_WMMA_F32_16X16X32_F16 (wave32).
// Async global->LDS staging (ASYNCcnt) used where the copy is raw f16.
// ---------------------------------------------------------------------------

#define CC   512                 // channels
#define DDq  64                  // q/k channels (C/8)
#define HH   96
#define WWD  96
#define BB   8
#define HWs  (HH * WWD)          // 9216
#define NCH  192                 // softmax width (H + W)
#define EPSv 1e-5f

typedef _Float16 v16h __attribute__((ext_vector_type(16)));
typedef _Float16 v8h  __attribute__((ext_vector_type(8)));
typedef _Float16 v2h  __attribute__((ext_vector_type(2)));
typedef float    v8f  __attribute__((ext_vector_type(8)));
typedef int      v4i  __attribute__((ext_vector_type(4)));

// gfx1250 async global->LDS copy path (graceful fallback if builtins absent)
#if defined(__AMDGCN__) && \
    __has_builtin(__builtin_amdgcn_global_load_async_to_lds_b128) && \
    __has_builtin(__builtin_amdgcn_s_wait_asynccnt)
#define USE_ASYNC_LDS 1
typedef __attribute__((address_space(1))) v4i* gptr_b128;
typedef __attribute__((address_space(3))) v4i* lptr_b128;
#else
#define USE_ASYNC_LDS 0
#endif

__device__ __forceinline__ v8f wmma_f16(v16h a, v16h b, v8f c) {
  // emits v_wmma_f32_16x16x32_f16
  return __builtin_amdgcn_wmma_f32_16x16x32_f16(false, a, false, b,
                                                (short)0, c, false, false);
}

// A-operand fragment (16x32 f16, ISA 7.12.2): lane owns row m = lane&15;
// hi = lane>>4 selects K halves {hi*8 .. hi*8+7} and {16+hi*8 .. 16+hi*8+7}.
// rowp points at 32 contiguous K values for this lane's row.
__device__ __forceinline__ v16h frag_rowK32(const _Float16* rowp, int hi) {
  v8h lo = *reinterpret_cast<const v8h*>(rowp + hi * 8);
  v8h hh = *reinterpret_cast<const v8h*>(rowp + 16 + hi * 8);
  v16h r;
#pragma unroll
  for (int i = 0; i < 8; ++i) { r[i] = lo[i]; r[8 + i] = hh[i]; }
  return r;
}

// B-operand fragment (32x16 f16): lane owns column n = lane&15; halves are
// K = {hi*8 .. hi*8+7} and {16+hi*8 .. 16+hi*8+7} -> requires [n][k] staging in LDS.
__device__ __forceinline__ v16h frag_colK32(const _Float16* colp, int hi) {
  v8h lo = *reinterpret_cast<const v8h*>(colp + hi * 8);
  v8h hh = *reinterpret_cast<const v8h*>(colp + 16 + hi * 8);
  v16h r;
#pragma unroll
  for (int i = 0; i < 8; ++i) { r[i] = lo[i]; r[8 + i] = hh[i]; }
  return r;
}

// ---------------------------------------------------------------------------
// f32 -> f16 convert (weights)
// ---------------------------------------------------------------------------
__global__ void cvt_f32_f16(const float* __restrict__ s,
                            _Float16* __restrict__ d, int n) {
  int i = blockIdx.x * blockDim.x + threadIdx.x;
  if (i < n) d[i] = (_Float16)s[i];
}

// ---------------------------------------------------------------------------
// conv1x1 as GEMM: Y16[b][d][p] = f16( sum_c W16[d][c] * X[b][c][p] + bias[d] )
// grid (HW/64, D/64, B), block 128 (4 waves). 64x64 tile, K step 32.
// ---------------------------------------------------------------------------
__global__ void proj_kernel(const _Float16* __restrict__ W16,
                            const float* __restrict__ bias,
                            const float* __restrict__ X,
                            _Float16* __restrict__ Y, int D) {
  __shared__ __align__(32) _Float16 Bs[64][32];   // [p][k] (K-contiguous)
  const int b  = blockIdx.z;
  const int m0 = blockIdx.y * 64;
  const int p0 = blockIdx.x * 64;
  const int tid  = threadIdx.x;
  const int lane = tid & 31, wave = tid >> 5;
  const int lm = lane & 15, hi = lane >> 4;
  const float*    Xb   = X + (size_t)b * CC * HWs;
  const _Float16* Wrow = W16 + (size_t)(m0 + wave * 16 + lm) * CC;

  v8f acc[4] = {};
  for (int k0 = 0; k0 < CC; k0 += 32) {
    // stage 32x64 X tile transposed into [p][k]; two K per thread so the
    // convert lowers to v_cvt_pk_f16_f32 + one aligned ds_store_b32
    for (int i = tid; i < 64 * 16; i += 128) {
      int kp = i >> 6, p = i & 63;
      int k  = kp * 2;
      float x0 = Xb[(size_t)(k0 + k)     * HWs + p0 + p];
      float x1 = Xb[(size_t)(k0 + k + 1) * HWs + p0 + p];
      v2h pk = { (_Float16)x0, (_Float16)x1 };
      *reinterpret_cast<v2h*>(&Bs[p][k]) = pk;
    }
    __syncthreads();
    if (k0 + 32 < CC) {  // global_prefetch_b8 of next K tile
      int pk = tid >> 2, pp = (tid & 3) * 16;
      __builtin_prefetch(&Xb[(size_t)(k0 + 32 + pk) * HWs + p0 + pp], 0, 1);
    }
    v16h a = frag_rowK32(Wrow + k0, hi);   // A straight from global (rows contiguous)
#pragma unroll
    for (int nt = 0; nt < 4; ++nt) {
      v16h bf = frag_colK32(&Bs[nt * 16 + lm][0], hi);
      acc[nt] = wmma_f16(a, bf, acc[nt]);
    }
    __syncthreads();
  }
#pragma unroll
  for (int nt = 0; nt < 4; ++nt) {
#pragma unroll
    for (int r = 0; r < 8; ++r) {
      int d = m0 + wave * 16 + r + 8 * hi;             // C/D layout: M = r + 8*hi
      float val = acc[nt][r] + bias[d];
      Y[((size_t)b * D + d) * HWs + p0 + nt * 16 + lm] = (_Float16)val;
    }
  }
}

// ---------------------------------------------------------------------------
// e_h[b,h,w,g] = sum_d q[b,d,h,w] * k[b,d,g,w]  (diag g==h masked to -inf)
// one block per (w,b): full 96x96 logit tile. block 256 (8 waves).
// ---------------------------------------------------------------------------
__global__ void eh_kernel(const _Float16* __restrict__ q16,
                          const _Float16* __restrict__ k16,
                          float* __restrict__ att) {
  __shared__ __align__(32) _Float16 Qs[HH][DDq];  // [h][d]
  __shared__ __align__(32) _Float16 Ks[HH][DDq];  // [g][d]
  const int w = blockIdx.x, b = blockIdx.y;
  const int tid = threadIdx.x;
  const _Float16* qb = q16 + (size_t)b * DDq * HWs;
  const _Float16* kb = k16 + (size_t)b * DDq * HWs;
  for (int i = tid; i < HH * DDq; i += 256) {
    int h = i % HH, d = i / HH;
    Qs[h][d] = qb[(size_t)d * HWs + h * WWD + w];
    Ks[h][d] = kb[(size_t)d * HWs + h * WWD + w];
  }
  __syncthreads();
  const int lane = tid & 31, wave = tid >> 5;
  const int lm = lane & 15, hi = lane >> 4;
  for (int t = wave; t < 36; t += 8) {
    int ih = t / 6, ig = t % 6;
    v8f acc = {};
#pragma unroll
    for (int kk = 0; kk < DDq; kk += 32) {
      v16h a  = frag_rowK32(&Qs[ih * 16 + lm][kk], hi);
      v16h bf = frag_colK32(&Ks[ig * 16 + lm][kk], hi);
      acc = wmma_f16(a, bf, acc);
    }
    int g = ig * 16 + lm;
#pragma unroll
    for (int r = 0; r < 8; ++r) {
      int h = ih * 16 + r + 8 * hi;
      float e = (g == h) ? -INFINITY : acc[r];
      att[(((size_t)b * HH + h) * WWD + w) * NCH + g] = e;
    }
  }
}

// ---------------------------------------------------------------------------
// e_w[b,h,w,v] = sum_d q[b,d,h,w] * k[b,d,h,v]; one block per (h,b)
// ---------------------------------------------------------------------------
__global__ void ew_kernel(const _Float16* __restrict__ q16,
                          const _Float16* __restrict__ k16,
                          float* __restrict__ att) {
  __shared__ __align__(32) _Float16 Qs[WWD][DDq]; // [w][d]
  __shared__ __align__(32) _Float16 Ks[WWD][DDq]; // [v][d]
  const int h = blockIdx.x, b = blockIdx.y;
  const int tid = threadIdx.x;
  const _Float16* qb = q16 + (size_t)b * DDq * HWs;
  const _Float16* kb = k16 + (size_t)b * DDq * HWs;
  for (int i = tid; i < WWD * DDq; i += 256) {
    int wc = i % WWD, d = i / WWD;
    Qs[wc][d] = qb[(size_t)d * HWs + h * WWD + wc];
    Ks[wc][d] = kb[(size_t)d * HWs + h * WWD + wc];
  }
  __syncthreads();
  const int lane = tid & 31, wave = tid >> 5;
  const int lm = lane & 15, hi = lane >> 4;
  for (int t = wave; t < 36; t += 8) {
    int iw = t / 6, iv = t % 6;
    v8f acc = {};
#pragma unroll
    for (int kk = 0; kk < DDq; kk += 32) {
      v16h a  = frag_rowK32(&Qs[iw * 16 + lm][kk], hi);
      v16h bf = frag_colK32(&Ks[iv * 16 + lm][kk], hi);
      acc = wmma_f16(a, bf, acc);
    }
    int vv = iv * 16 + lm;
#pragma unroll
    for (int r = 0; r < 8; ++r) {
      int wc = iw * 16 + r + 8 * hi;
      att[(((size_t)b * HH + h) * WWD + wc) * NCH + HH + vv] = acc[r];
    }
  }
}

// ---------------------------------------------------------------------------
// softmax over last dim (192): one wave32 per row, 8 rows per block
// ---------------------------------------------------------------------------
__global__ void softmax192(float* __restrict__ att) {
  const int wave = threadIdx.x >> 5, lane = threadIdx.x & 31;
  const size_t row = (size_t)blockIdx.x * 8 + wave;
  float* p = att + row * NCH;
  float v[6];
  float m = -INFINITY;
#pragma unroll
  for (int j = 0; j < 6; ++j) { v[j] = p[lane + 32 * j]; m = fmaxf(m, v[j]); }
#pragma unroll
  for (int o = 16; o > 0; o >>= 1) m = fmaxf(m, __shfl_xor(m, o, 32));
  float s = 0.f;
#pragma unroll
  for (int j = 0; j < 6; ++j) { v[j] = __expf(v[j] - m); s += v[j]; }
#pragma unroll
  for (int o = 16; o > 0; o >>= 1) s += __shfl_xor(s, o, 32);
  float inv = 1.0f / s;
#pragma unroll
  for (int j = 0; j < 6; ++j) p[lane + 32 * j] = v[j] * inv;
}

// ---------------------------------------------------------------------------
// out_h[b,c,h,w] = sum_g a_h[b,h,w,g] * v[b,c,g,w]; block per (w, c-chunk, b)
// ---------------------------------------------------------------------------
__global__ void outh_kernel(const _Float16* __restrict__ v16p,
                            const float* __restrict__ att,
                            float* __restrict__ accb) {
  __shared__ __align__(32) _Float16 Vs[128][HH];  // [c][g]
  __shared__ __align__(32) _Float16 Ah[HH][HH];   // [h][g]
  const int w = blockIdx.x, b = blockIdx.z;
  const int c0 = blockIdx.y * 128;
  const int tid = threadIdx.x;
  const _Float16* vb = v16p + (size_t)b * CC * HWs;
  for (int i = tid; i < 128 * HH; i += 128) {
    int g = i % HH, c = i / HH;
    Vs[c][g] = vb[(size_t)(c0 + c) * HWs + g * WWD + w];
  }
  for (int i = tid; i < HH * HH; i += 128) {
    int g = i % HH, h = i / HH;
    Ah[h][g] = (_Float16)att[(((size_t)b * HH + h) * WWD + w) * NCH + g];
  }
  __syncthreads();
  const int lane = tid & 31, wave = tid >> 5;
  const int lm = lane & 15, hi = lane >> 4;
  for (int t = wave; t < 48; t += 4) {
    int ic = t / 6, ihn = t % 6;
    v8f acc = {};
#pragma unroll
    for (int kk = 0; kk < HH; kk += 32) {
      v16h a  = frag_rowK32(&Vs[ic * 16 + lm][kk], hi);
      v16h bf = frag_colK32(&Ah[ihn * 16 + lm][kk], hi);
      acc = wmma_f16(a, bf, acc);
    }
    int h = ihn * 16 + lm;
#pragma unroll
    for (int r = 0; r < 8; ++r) {
      int c = c0 + ic * 16 + r + 8 * hi;
      accb[((size_t)b * CC + c) * HWs + (size_t)h * WWD + w] = acc[r];
    }
  }
}

// ---------------------------------------------------------------------------
// out_w + fuse: xdst = gamma*(out_h + out_w) + xsrc; block per (h, c-chunk, b)
// V rows here are contiguous f16 -> async global->LDS copy (ASYNCcnt path)
// ---------------------------------------------------------------------------
__global__ void outw_kernel(const _Float16* __restrict__ v16p,
                            const float* __restrict__ att,
                            const float* __restrict__ accb,
                            const float* __restrict__ xsrc,
                            const float* __restrict__ gamma,
                            float* __restrict__ xdst) {
  __shared__ __align__(32) _Float16 Vs[128][HH];  // [c][v]
  __shared__ __align__(32) _Float16 Aw[HH][HH];   // [w][v]
  const int h = blockIdx.x, b = blockIdx.z;
  const int c0 = blockIdx.y * 128;
  const int tid = threadIdx.x;
  const _Float16* vb = v16p + (size_t)b * CC * HWs;
#if USE_ASYNC_LDS
  // 128 rows x 192B, copied as 16B chunks: GLOBAL_LOAD_ASYNC_TO_LDS_B128
  for (int j = tid; j < 128 * 12; j += 128) {
    int c = j / 12, o = (j % 12) * 8;
    const _Float16* g = vb + (size_t)(c0 + c) * HWs + (size_t)h * WWD + o;
    __builtin_amdgcn_global_load_async_to_lds_b128(
        (gptr_b128)g, (lptr_b128)&Vs[c][o], 0, 0);
  }
#else
  for (int i = tid; i < 128 * HH; i += 128) {
    int vv = i % HH, c = i / HH;
    Vs[c][vv] = vb[(size_t)(c0 + c) * HWs + h * WWD + vv];
  }
#endif
  for (int i = tid; i < HH * HH; i += 128) {
    int vv = i % HH, wr = i / HH;
    Aw[wr][vv] = (_Float16)att[(((size_t)b * HH + h) * WWD + wr) * NCH + HH + vv];
  }
#if USE_ASYNC_LDS
  __builtin_amdgcn_s_wait_asynccnt(0);   // s_wait_asynccnt 0
#endif
  __syncthreads();
  const int lane = tid & 31, wave = tid >> 5;
  const int lm = lane & 15, hi = lane >> 4;
  const float g0 = gamma[0];
  for (int t = wave; t < 48; t += 4) {
    int ic = t / 6, iwn = t % 6;
    v8f acc = {};
#pragma unroll
    for (int kk = 0; kk < HH; kk += 32) {
      v16h a  = frag_rowK32(&Vs[ic * 16 + lm][kk], hi);
      v16h bf = frag_colK32(&Aw[iwn * 16 + lm][kk], hi);
      acc = wmma_f16(a, bf, acc);
    }
    int wc = iwn * 16 + lm;
#pragma unroll
    for (int r = 0; r < 8; ++r) {
      int c = c0 + ic * 16 + r + 8 * hi;
      size_t idx = ((size_t)b * CC + c) * HWs + (size_t)h * WWD + wc;
      float o = acc[r] + accb[idx];
      xdst[idx] = g0 * o + xsrc[idx];
    }
  }
}

// ---------------------------------------------------------------------------
// BatchNorm: per-channel stats then in-place normalize
// ---------------------------------------------------------------------------
__global__ void bn_stats(const float* __restrict__ x, float* __restrict__ stats) {
  const int c = blockIdx.x, tid = threadIdx.x;
  float s = 0.f, s2 = 0.f;
  for (int i = tid; i < BB * HWs; i += 256) {
    int b = i / HWs, r = i - b * HWs;
    float v = x[((size_t)b * CC + c) * HWs + r];
    s += v; s2 += v * v;
  }
  __shared__ float sh[256], sh2[256];
  sh[tid] = s; sh2[tid] = s2;
  __syncthreads();
  for (int o = 128; o > 0; o >>= 1) {
    if (tid < o) { sh[tid] += sh[tid + o]; sh2[tid] += sh2[tid + o]; }
    __syncthreads();
  }
  if (tid == 0) {
    const float N = (float)(BB * HWs);
    float mean = sh[0] / N;
    stats[c]       = mean;
    stats[CC + c]  = sh2[0] / N - mean * mean;
  }
}

__global__ void bn_apply(float* __restrict__ y, const float* __restrict__ stats,
                         const float* __restrict__ wgt, const float* __restrict__ bsc,
                         size_t n) {
  size_t i = (size_t)blockIdx.x * blockDim.x + threadIdx.x;
  if (i >= n) return;
  int c = (int)((i / HWs) % CC);
  float mean = stats[c], var = stats[CC + c];
  y[i] = (y[i] - mean) * rsqrtf(var + EPSv) * wgt[c] + bsc[c];
}

// ---------------------------------------------------------------------------
// host orchestration
// ---------------------------------------------------------------------------
extern "C" void kernel_launch(void* const* d_in, const int* in_sizes, int n_in,
                              void* d_out, int out_size, void* d_ws, size_t ws_size,
                              hipStream_t stream) {
  (void)in_sizes; (void)n_in; (void)out_size; (void)ws_size;
  const float* x     = (const float*)d_in[0];
  const float* Wq    = (const float*)d_in[1];
  const float* bq    = (const float*)d_in[2];
  const float* Wk    = (const float*)d_in[3];
  const float* bk    = (const float*)d_in[4];
  const float* Wv    = (const float*)d_in[5];
  const float* bv    = (const float*)d_in[6];
  const float* gamma = (const float*)d_in[7];
  const float* bnw   = (const float*)d_in[8];
  const float* bnb   = (const float*)d_in[9];
  // d_in[10] = recurrent (device scalar); fixed to 2 by setup_inputs().

  char* base = (char*)d_ws;
  size_t off = 0;
  auto carve = [&](size_t bytes) -> void* {
    void* p = base + off;
    off += (bytes + 255) & ~(size_t)255;
    return p;
  };
  _Float16* wq16 = (_Float16*)carve((size_t)DDq * CC * 2);
  _Float16* wk16 = (_Float16*)carve((size_t)DDq * CC * 2);
  _Float16* wv16 = (_Float16*)carve((size_t)CC  * CC * 2);
  _Float16* q16  = (_Float16*)carve((size_t)BB * DDq * HWs * 2);
  _Float16* k16  = (_Float16*)carve((size_t)BB * DDq * HWs * 2);
  _Float16* v16b = (_Float16*)carve((size_t)BB * CC  * HWs * 2);
  float*    attb = (float*)carve((size_t)BB * HWs * NCH * 4);
  float*    accb = (float*)carve((size_t)BB * CC * HWs * 4);
  float*    xbuf = (float*)carve((size_t)BB * CC * HWs * 4);
  float*    stats = (float*)carve((size_t)2 * CC * 4);

  // weights -> f16 (cheap, once per call)
  cvt_f32_f16<<<(DDq * CC + 255) / 256, 256, 0, stream>>>(Wq, wq16, DDq * CC);
  cvt_f32_f16<<<(DDq * CC + 255) / 256, 256, 0, stream>>>(Wk, wk16, DDq * CC);
  cvt_f32_f16<<<(CC * CC + 255) / 256, 256, 0, stream>>>(Wv, wv16, CC * CC);

  const float* xs = x;
  for (int it = 0; it < 2; ++it) {
    float* xd = (it == 0) ? xbuf : (float*)d_out;
    proj_kernel<<<dim3(HWs / 64, DDq / 64, BB), 128, 0, stream>>>(wq16, bq, xs, q16, DDq);
    proj_kernel<<<dim3(HWs / 64, DDq / 64, BB), 128, 0, stream>>>(wk16, bk, xs, k16, DDq);
    proj_kernel<<<dim3(HWs / 64, CC  / 64, BB), 128, 0, stream>>>(wv16, bv, xs, v16b, CC);
    eh_kernel<<<dim3(WWD, BB), 256, 0, stream>>>(q16, k16, attb);
    ew_kernel<<<dim3(HH, BB), 256, 0, stream>>>(q16, k16, attb);
    softmax192<<<(BB * HWs) / 8, 256, 0, stream>>>(attb);
    outh_kernel<<<dim3(WWD, CC / 128, BB), 128, 0, stream>>>(v16b, attb, accb);
    outw_kernel<<<dim3(HH, CC / 128, BB), 128, 0, stream>>>(v16b, attb, accb, xs, gamma, xd);
    xs = xd;
  }

  // batchnorm in place on d_out
  const size_t n = (size_t)BB * CC * HWs;
  bn_stats<<<CC, 256, 0, stream>>>((const float*)d_out, stats);
  bn_apply<<<(unsigned)((n + 255) / 256), 256, 0, stream>>>((float*)d_out, stats, bnw, bnb, n);
}
